// self_attention_18167711662780
// MI455X (gfx1250) — matrix operand order repaired and checked by reading, not compiled, not run on hardware
//
#include <hip/hip_runtime.h>

#define DEV __device__ __forceinline__

typedef _Float16 v8h  __attribute__((ext_vector_type(8)));
typedef _Float16 v16h __attribute__((ext_vector_type(16)));
typedef float    v8f  __attribute__((ext_vector_type(8)));

static constexpr int B_  = 4;
static constexpr int S_  = 2048;
static constexpr int C_  = 1024;
static constexpr int DK_ = 256;
static constexpr int DO_ = 1024;

// ---------------- conversion kernels ----------------

__global__ void cvt_f32_to_f16(const float* __restrict__ in, _Float16* __restrict__ out, int n) {
    int i = blockIdx.x * blockDim.x + threadIdx.x;
    int stride = gridDim.x * blockDim.x;
    for (; i < n; i += stride) out[i] = (_Float16)in[i];
}

// W: [R][N] row-major f32  ->  Wt: [N][R] row-major f16
__global__ void transpose_cvt(const float* __restrict__ W, _Float16* __restrict__ Wt, int R, int N) {
    int i = blockIdx.x * blockDim.x + threadIdx.x;
    if (i >= R * N) return;
    int r = i / N, n = i % N;
    Wt[(size_t)n * R + r] = (_Float16)W[i];
}

// ---------------- WMMA fragment loaders ----------------
// A fragment (16x32 f16, M=lane%16): VGPR0-7 hold K = half*16 + 0..15
// (half = lane/16).
DEV v16h load_a_frag(const _Float16* rowk, int half) {
    v8h c0 = *(const v8h*)(rowk + half * 16);
    v8h c1 = *(const v8h*)(rowk + half * 16 + 8);
    v16h r;
#pragma unroll
    for (int i = 0; i < 8; i++) { r[i] = c0[i]; r[i + 8] = c1[i]; }
    return r;
}

// B fragment (32x16 f16) from N-major (transposed) storage: lane holds
// column n=lane%16; 16 contiguous K values starting at khalf*16.
// Caller passes pointer already offset to (row n, k0 + khalf*16).
DEV v16h load_b_frag(const _Float16* p) {
    v8h c0 = *(const v8h*)(p);
    v8h c1 = *(const v8h*)(p + 8);
    v16h r;
#pragma unroll
    for (int i = 0; i < 8; i++) { r[i] = c0[i]; r[i + 8] = c1[i]; }
    return r;
}

DEV v8f wmma_f16(v16h a, v16h b, v8f c) {
    return __builtin_amdgcn_wmma_f32_16x16x32_f16(false, a, false, b, (short)0, c,
                                                  false, false);
}

// ---------------- QKV projection GEMM ----------------
// Y = Xh(8192 x 1024) * W(1024 x N) + bias ; W given transposed f16 [N][1024].
// Wave computes a 16x64 output tile. Two-phase register rotation: fragment
// sets (a0,b0)/(a1,b1) alternate, prefetch runs 2 k-steps ahead, and no
// register-copy rotation is required.
__global__ __launch_bounds__(256) void gemm_proj(
    const _Float16* __restrict__ Xh, const _Float16* __restrict__ Wt,
    const float* __restrict__ bias, _Float16* __restrict__ out,
    int N, int transpose_out) {
    const int tid  = threadIdx.x;
    const int lane = tid & 31, wv = tid >> 5;
    const int gw   = blockIdx.x * 8 + wv;
    const int wavesPerRow = N >> 6;
    const int mt   = gw / wavesPerRow;
    const int col0 = (gw % wavesPerRow) << 6;
    const int m16  = lane & 15, half = lane >> 4;

    const _Float16* arow = Xh + (size_t)(mt * 16 + m16) * C_;
    const _Float16* brow = Wt + (size_t)(col0 + m16) * C_ + half * 16;

    v8f acc[4] = {};

    auto load_set = [&](int k0, v16h& a, v16h (&b)[4]) {
        a = load_a_frag(arow + k0, half);
#pragma unroll
        for (int t = 0; t < 4; t++)
            b[t] = load_b_frag(brow + (size_t)t * 16 * C_ + k0);
    };
    auto do_wmma = [&](const v16h& a, const v16h (&b)[4]) {
#pragma unroll
        for (int t = 0; t < 4; t++) acc[t] = wmma_f16(a, b[t], acc[t]);
    };

    v16h a0, a1;
    v16h b0[4], b1[4];
    load_set(0, a0, b0);     // k-step 0
    load_set(32, a1, b1);    // k-step 1

    for (int k0 = 64; k0 < C_; k0 += 64) {
        do_wmma(a0, b0);
        load_set(k0, a0, b0);       // prefetch 2 steps ahead of this wmma
        do_wmma(a1, b1);
        load_set(k0 + 32, a1, b1);
    }
    do_wmma(a0, b0);
    do_wmma(a1, b1);

#pragma unroll
    for (int t = 0; t < 4; t++) {
        int col  = col0 + 16 * t + m16;       // C/D layout: lane%16 = column
        float bv = bias[col];
#pragma unroll
        for (int r = 0; r < 8; r++) {
            int rg  = mt * 16 + r + 8 * half; // C/D layout: row = vgpr + 8*(lane/16)
            float v = acc[t][r] + bv;
            size_t idx;
            if (transpose_out) {
                int bbi = rg / S_, s = rg % S_;
                idx = (size_t)bbi * DO_ * S_ + (size_t)col * S_ + s;
            } else {
                idx = (size_t)rg * N + col;
            }
            out[idx] = (_Float16)v;
        }
    }
}

// ---------------- flash attention ----------------
// Block = 256 threads (8 waves) handles one (batch, 16 query rows).
// Wave w owns DO columns [w*128, w*128+128). Per 64-key block:
//   wave w computes the 16x16 score tile for keys [k0+16w, k0+16w+16)
//   -> online softmax (shfl + LDS reductions) -> P tile (16x64 f16) in LDS
//   -> each wave: 2 A-frags of P x 8 B-frags of V^T (L2) = 16 WMMAs, pipelined.
// Inner loops are fully unrolled, so SSA renaming gives copy-free pipelining.
__global__ __launch_bounds__(256) void flash_attn(
    const _Float16* __restrict__ Qh, const _Float16* __restrict__ Kh,
    const _Float16* __restrict__ Vt, float* __restrict__ Out) {
    const int tid  = threadIdx.x;
    const int lane = tid & 31, w = tid >> 5;
    const int m16  = lane & 15, half = lane >> 4;
    const int bb   = blockIdx.x / (S_ / 16);
    const int q0   = (blockIdx.x % (S_ / 16)) * 16;

    __shared__ _Float16 Psh[16 * 64];
    __shared__ float red[8][16];
    __shared__ float mrow[16], lrow[16], alpha_s[16];

    if (tid < 16) { mrow[tid] = -1e30f; lrow[tid] = 0.0f; }
    __syncthreads();

    // Q tile 16x256 held in registers as 8 A-fragments (64 VGPRs)
    const _Float16* qrow = Qh + (size_t)(bb * S_ + q0 + m16) * DK_;
    v16h qf[8];
#pragma unroll
    for (int s = 0; s < 8; s++) qf[s] = load_a_frag(qrow + s * 32, half);

    const int colbase = w * 128;
    // per-wave V^T base: row = output column (d), contiguous k (= key index)
    const _Float16* vbase =
        Vt + (size_t)(bb * DO_ + colbase + m16) * S_ + half * 16;
    const _Float16* krow0 =
        Kh + (size_t)(bb * S_ + w * 16 + m16) * DK_ + half * 16;

    v8f acc[8] = {};
    const float sc = 0.0625f;  // 1/sqrt(DK)

    for (int j = 0; j < S_ / 64; j++) {
        const int k0 = j * 64;

        // ---- scores: Q(16x256) x K_block^T, pipelined over the 8 d-steps
        const _Float16* krow = krow0 + (size_t)k0 * DK_;
        v8f sa = {};
        v16h kb = load_b_frag(krow);
#pragma unroll
        for (int s = 0; s < 7; s++) {
            v16h kb_n = load_b_frag(krow + (s + 1) * 32);
            sa = wmma_f16(qf[s], kb, sa);
            kb = kb_n;
        }
        sa = wmma_f16(qf[7], kb, sa);

        // per-row max within this wave's 16 columns (reduce across 16-lane half)
        float sval[8], rm[8];
#pragma unroll
        for (int r = 0; r < 8; r++) {
            float v = sa[r] * sc;
            sval[r] = v;
            float mx = v;
            mx = fmaxf(mx, __shfl_xor(mx, 1, 32));
            mx = fmaxf(mx, __shfl_xor(mx, 2, 32));
            mx = fmaxf(mx, __shfl_xor(mx, 4, 32));
            mx = fmaxf(mx, __shfl_xor(mx, 8, 32));
            rm[r] = mx;
        }
        if (lane == 0) {
            for (int r = 0; r < 8; r++) red[w][r] = rm[r];
        } else if (lane == 16) {
            for (int r = 0; r < 8; r++) red[w][8 + r] = rm[r];
        }
        __syncthreads();

        // ---- global row max / rescale factor
        if (tid < 16) {
            float gm = red[0][tid];
#pragma unroll
            for (int u = 1; u < 8; u++) gm = fmaxf(gm, red[u][tid]);
            float mo = mrow[tid];
            float mn = fmaxf(mo, gm);
            float a  = __expf(mo - mn);
            alpha_s[tid] = a;
            lrow[tid] *= a;
            mrow[tid] = mn;
        }
        __syncthreads();

        // ---- P = exp(S - m), stash f16 tile in LDS, row sums
        float rs[8];
#pragma unroll
        for (int r = 0; r < 8; r++) {
            int rr = r + 8 * half;
            float pv = __expf(sval[r] - mrow[rr]);
            Psh[rr * 64 + w * 16 + m16] = (_Float16)pv;
            float s = pv;
            s += __shfl_xor(s, 1, 32);
            s += __shfl_xor(s, 2, 32);
            s += __shfl_xor(s, 4, 32);
            s += __shfl_xor(s, 8, 32);
            rs[r] = s;
        }
        if (lane == 0) {
            for (int r = 0; r < 8; r++) red[w][r] = rs[r];
        } else if (lane == 16) {
            for (int r = 0; r < 8; r++) red[w][8 + r] = rs[r];
        }
        __syncthreads();

        if (tid < 16) {
            float l = lrow[tid];
#pragma unroll
            for (int u = 0; u < 8; u++) l += red[u][tid];
            lrow[tid] = l;
        }

        // ---- rescale accumulators
        float al[8];
#pragma unroll
        for (int r = 0; r < 8; r++) al[r] = alpha_s[r + 8 * half];
#pragma unroll
        for (int t = 0; t < 8; t++)
#pragma unroll
            for (int r = 0; r < 8; r++) acc[t][r] *= al[r];

        // ---- acc += P(16x64) x V_block, 16 WMMAs pipelined one B-frag ahead
        {
            v16h af0 = load_a_frag(&Psh[m16 * 64], half);
            v16h af1 = load_a_frag(&Psh[m16 * 64 + 32], half);
            v16h bf = load_b_frag(vbase + k0);
#pragma unroll
            for (int i = 0; i < 16; i++) {
                const int kk = i >> 3, t = i & 7;
                v16h bf_n;
                if (i < 15) {
                    const int kkn = (i + 1) >> 3, tn = (i + 1) & 7;
                    bf_n = load_b_frag(vbase + (size_t)tn * 16 * S_ + k0 + kkn * 32);
                }
                acc[t] = wmma_f16(kk ? af1 : af0, bf, acc[t]);
                bf = bf_n;
            }
        }
        __syncthreads();
    }

    // ---- normalize and store f32 output
    float linv[8];
#pragma unroll
    for (int r = 0; r < 8; r++) linv[r] = 1.0f / lrow[r + 8 * half];
#pragma unroll
    for (int t = 0; t < 8; t++) {
        int col = colbase + 16 * t + m16;
#pragma unroll
        for (int r = 0; r < 8; r++) {
            size_t idx = (size_t)(bb * S_ + q0 + r + 8 * half) * DO_ + col;
            Out[idx] = acc[t][r] * linv[r];
        }
    }
}

// ---------------- host side ----------------

extern "C" void kernel_launch(void* const* d_in, const int* in_sizes, int n_in,
                              void* d_out, int out_size, void* d_ws, size_t ws_size,
                              hipStream_t stream) {
    const float* x  = (const float*)d_in[0];
    const float* Wq = (const float*)d_in[1];
    const float* bq = (const float*)d_in[2];
    const float* Wk = (const float*)d_in[3];
    const float* bk = (const float*)d_in[4];
    const float* Wv = (const float*)d_in[5];
    const float* bv = (const float*)d_in[6];
    float* out = (float*)d_out;

    char* ws = (char*)d_ws;
    size_t off = 0;
    auto alloc = [&](size_t bytes) -> char* {
        char* p = ws + off;
        off += (bytes + 255) & ~(size_t)255;
        return p;
    };
    _Float16* xh  = (_Float16*)alloc((size_t)B_ * S_ * C_ * 2);
    _Float16* wqt = (_Float16*)alloc((size_t)DK_ * C_ * 2);
    _Float16* wkt = (_Float16*)alloc((size_t)DK_ * C_ * 2);
    _Float16* wvt = (_Float16*)alloc((size_t)DO_ * C_ * 2);
    _Float16* qh  = (_Float16*)alloc((size_t)B_ * S_ * DK_ * 2);
    _Float16* kh  = (_Float16*)alloc((size_t)B_ * S_ * DK_ * 2);
    _Float16* vt  = (_Float16*)alloc((size_t)B_ * S_ * DO_ * 2);  // [b][d][s]

    const int nx = B_ * S_ * C_;
    cvt_f32_to_f16<<<4096, 256, 0, stream>>>(x, xh, nx);
    transpose_cvt<<<(C_ * DK_ + 255) / 256, 256, 0, stream>>>(Wq, wqt, C_, DK_);
    transpose_cvt<<<(C_ * DK_ + 255) / 256, 256, 0, stream>>>(Wk, wkt, C_, DK_);
    transpose_cvt<<<(C_ * DO_ + 255) / 256, 256, 0, stream>>>(Wv, wvt, C_, DO_);

    const int M = B_ * S_;
    gemm_proj<<<(M / 16) * (DK_ / 64) / 8, 256, 0, stream>>>(xh, wqt, bq, qh, DK_, 0);
    gemm_proj<<<(M / 16) * (DK_ / 64) / 8, 256, 0, stream>>>(xh, wkt, bk, kh, DK_, 0);
    gemm_proj<<<(M / 16) * (DO_ / 64) / 8, 256, 0, stream>>>(xh, wvt, bv, vt, DO_, 1);

    flash_attn<<<B_ * (S_ / 16), 256, 0, stream>>>(qh, kh, vt, out);
}
